// Metabolism_Propagation_29411936043039
// MI455X (gfx1250) — matrix operand order repaired and checked by reading, not compiled, not run on hardware
//
#include <hip/hip_runtime.h>
#include <math.h>

#define H 64

typedef _Float16 half8 __attribute__((ext_vector_type(8)));
typedef _Float16 v16h  __attribute__((ext_vector_type(16)));
typedef float    v8f   __attribute__((ext_vector_type(8)));
typedef float    f4    __attribute__((ext_vector_type(4)));

union V16H { v16h v; half8 h8[2]; _Float16 h[16]; };
union V8F  { v8f  v; float f[8]; };

#if defined(__has_builtin)
#  if __has_builtin(__builtin_amdgcn_tanhf)
#    define HAVE_HW_TANH 1
#  endif
#endif

// Branch-free tanh: hardware V_TANH_F32 when available, else exp2-based.
__device__ __forceinline__ float fast_tanh(float x) {
#ifdef HAVE_HW_TANH
    return __builtin_amdgcn_tanhf(x);
#else
    float ax = fabsf(x);
    float e  = __builtin_amdgcn_exp2f(ax * 2.88539008177793f);  // exp(2|x|)
    float t  = 1.0f - 2.0f / (e + 1.0f);
    return copysignf(t, x);
#endif
}

// ---------------------------------------------------------------------------
// Transposed formulation:  h1^T(64 x 16edges) = W1^T(64x64) @ h0^T(64x16)
//   A = W1^T tiles (registers, block-constant)
//   B = h0^T tiles: ISA 16-bit B layout => lane `col` holds two 8-runs
//       of h0[edge=col] (k = 8*hi16+j, then 16+8*hi16+j); layer 1 produces fragments directly in regs.
//   C: M = hidden row, N = edge column => layer-3 dot is in-lane; one
//      shfl_xor(16) finishes the 64-wide reduction.
//   MODE 0: edge messages  -> atomicAdd(agg[rxn_sub[g]], msg)
//   MODE 1: node homeostasis -> out[g] = msg
// ---------------------------------------------------------------------------
template<int MODE>
__global__ __launch_bounds__(256)
void mlp_kernel(const float* __restrict__ x,
                const float* __restrict__ a,
                const float* __restrict__ sto_all,
                const int*   __restrict__ met_sub,
                const int*   __restrict__ sub_to_all,
                const int*   __restrict__ rxn_sub,
                const float* __restrict__ W0, const float* __restrict__ b0,
                const float* __restrict__ W1, const float* __restrict__ b1,
                const float* __restrict__ W2, const float* __restrict__ b2,
                float* __restrict__ out, int n_items)
{
    __shared__ _Float16 sW1[H * H];                 // 8 KB f16 copy of 64x64 weights
    __shared__ alignas(16) float sPack[H][4];       // {b0[j], W0[0][j], W0[1][j], W0[2][j]|0}
    __shared__ alignas(16) float sB1[H];
    __shared__ alignas(16) float sW2v[H];

    const int tid  = threadIdx.x;
    const int lane = tid & 31;

    // ---- stage weights into LDS (once per block) ----
    for (int i = tid; i < H * H; i += 256) sW1[i] = (_Float16)W1[i];
    if (tid < H) {
        sPack[tid][0] = b0[tid];
        sPack[tid][1] = W0[0 * H + tid];
        sPack[tid][2] = W0[1 * H + tid];
        sPack[tid][3] = (MODE == 1) ? W0[2 * H + tid] : 0.f;
        sB1[tid]  = b1[tid];
        sW2v[tid] = W2[tid];
    }
    __syncthreads();
    const float bias2 = b2[0];

    const int col  = lane & 15;
    const int hi16 = lane >> 4;          // 0: lanes 0-15, 1: lanes 16-31

    // ---- pre-pack A fragments = W1^T per ISA 16-bit A layout (block-constant) ----
    // A tile (Mt,Kt): M = Mt*16+col ; halfs 0-7: K = Kt*32 + hi16*8 + j,
    //                                 halfs 8-15: K = Kt*32 + 16 + hi16*8 + j.
    V16H Wf[4][2];
    for (int Mt = 0; Mt < 4; ++Mt) {
        const int m  = Mt * 16 + col;
        const int hb = hi16 * 8;
        for (int Kt = 0; Kt < 2; ++Kt) {
            for (int j = 0; j < 8; ++j) {
                Wf[Mt][Kt].h[j]     = sW1[(Kt * 32 + hb + j) * H + m];       // W1^T[m][k]
                Wf[Mt][Kt].h[j + 8] = sW1[(Kt * 32 + 16 + hb + j) * H + m];
            }
        }
    }

    const int waveGlobal = blockIdx.x * 8 + (tid >> 5);
    const int nWaves     = gridDim.x * 8;
    const int nTiles     = (n_items + 15) >> 4;

    for (int tile = waveGlobal; tile < nTiles; tile += nWaves) {
        const int base = tile * 16;

        // ---- gather inputs for edge/node  m = col ----
        int g = base + col;
        if (g > n_items - 1) g = n_items - 1;       // clamp: keep EXEC full for WMMA
        float in0, in1, in2 = 0.f;
        if (MODE == 0) {
            in0 = x[met_sub[g] * 5 + 3];
            in1 = fabsf(sto_all[sub_to_all[g]]);
        } else {
            in0 = x[g * 5 + 3];
            in1 = a[g * 2 + 0];
            in2 = a[g * 2 + 1];
        }

        // ---- layer 1 (VALU) directly into B fragments (ISA 16-bit B layout):
        //      lane holds h0[col][Kt*32 + (j<8 ? hi16*8 + j : 16 + hi16*8 + j-8)], j = 0..15
        V16H Act[2];
        for (int Kt = 0; Kt < 2; ++Kt) {
            const int jb = Kt * 32 + hi16 * 8;
            for (int j = 0; j < 16; ++j) {
                const f4 w = *(const f4*)&sPack[jb + (j < 8 ? j : j + 8)][0];   // one ds_load_b128
                float hsum = w[0] + in0 * w[1] + in1 * w[2];
                if (MODE == 1) hsum += in2 * w[3];
                Act[Kt].h[j] = (_Float16)fast_tanh(hsum);
            }
        }

        // ---- layer 2 on WMMA (M=hidden, N=edge), fused tanh + layer-3 dot.
        //      4 independent partial accumulators break the tanh->fmac chain.
        float ps[4] = {0.f, 0.f, 0.f, 0.f};
        for (int Mt = 0; Mt < 4; ++Mt) {
            const int rb = Mt * 16 + 8 * hi16;       // this lane's 8 hidden rows
            const f4 c0  = *(const f4*)&sB1[rb];
            const f4 c1  = *(const f4*)&sB1[rb + 4];
            v8f acc;
            for (int r = 0; r < 4; ++r) { acc[r] = c0[r]; acc[r + 4] = c1[r]; }
            acc = __builtin_amdgcn_wmma_f32_16x16x32_f16(false, Wf[Mt][0].v, false, Act[0].v,
                                                         (short)0, acc, false, false);
            acc = __builtin_amdgcn_wmma_f32_16x16x32_f16(false, Wf[Mt][1].v, false, Act[1].v,
                                                         (short)0, acc, false, false);
            const f4 w20 = *(const f4*)&sW2v[rb];
            const f4 w21 = *(const f4*)&sW2v[rb + 4];
            V8F av; av.v = acc;
            for (int r = 0; r < 4; ++r) {
                ps[r]  = fmaf(fast_tanh(av.f[r]),     w20[r], ps[r]);
            }
            for (int r = 0; r < 4; ++r) {
                ps[r]  = fmaf(fast_tanh(av.f[r + 4]), w21[r], ps[r]);
            }
        }
        float p = (ps[0] + ps[1]) + (ps[2] + ps[3]);

        // ---- single butterfly: combine lanes e and e+16 -> full 64-sum ----
        p += __shfl_xor(p, 16, 32);

        if (hi16 == 0) {
            const int ge = base + col;
            if (ge < n_items) {
                const float msg = p + bias2;
                if (MODE == 0) atomicAdd(&out[rxn_sub[ge]], msg);
                else           out[ge] = msg;
            }
        }
    }
}

// ---------------------------------------------------------------------------
__global__ void zero_kernel(float* __restrict__ p, int n) {
    int i = blockIdx.x * blockDim.x + threadIdx.x;
    if (i < n) p[i] = 0.f;
}

// v[r] = 10^log_k[r] * softplus(agg[r])   (in place over agg)
__global__ void rxn_kernel(const float* __restrict__ log_k, float* __restrict__ agg, int n) {
    int i = blockIdx.x * blockDim.x + threadIdx.x;
    if (i < n) {
        float g  = agg[i];
        float sp = (g > 20.f) ? g : log1pf(__expf(g));
        agg[i] = exp10f(log_k[i]) * sp;
    }
}

// dxdt[met_all[i]] += sto_all[i] * v[rxn_all[i]]
__global__ void contrib_kernel(const float* __restrict__ sto_all,
                               const float* __restrict__ v,
                               const int*   __restrict__ rxn_all,
                               const int*   __restrict__ met_all,
                               float* __restrict__ out, int n) {
    int i = blockIdx.x * blockDim.x + threadIdx.x;
    if (i < n) atomicAdd(&out[met_all[i]], sto_all[i] * v[rxn_all[i]]);
}

// ---------------------------------------------------------------------------
extern "C" void kernel_launch(void* const* d_in, const int* in_sizes, int n_in,
                              void* d_out, int out_size, void* d_ws, size_t ws_size,
                              hipStream_t stream) {
    const float* x          = (const float*)d_in[0];
    const float* a          = (const float*)d_in[1];
    const float* sto_all    = (const float*)d_in[2];
    const float* log_k      = (const float*)d_in[3];
    const float* sW0        = (const float*)d_in[4];
    const float* sb0        = (const float*)d_in[5];
    const float* sW1        = (const float*)d_in[6];
    const float* sb1        = (const float*)d_in[7];
    const float* sW2        = (const float*)d_in[8];
    const float* sb2        = (const float*)d_in[9];
    const float* nW0        = (const float*)d_in[10];
    const float* nb0        = (const float*)d_in[11];
    const float* nW1        = (const float*)d_in[12];
    const float* nb1        = (const float*)d_in[13];
    const float* nW2        = (const float*)d_in[14];
    const float* nb2        = (const float*)d_in[15];
    const int*   met_sub    = (const int*)d_in[16];
    const int*   rxn_sub    = (const int*)d_in[17];
    const int*   sub_to_all = (const int*)d_in[18];
    const int*   met_all    = (const int*)d_in[19];
    const int*   rxn_all    = (const int*)d_in[20];

    const int n_met = in_sizes[0] / 5;
    const int n_all = in_sizes[2];
    const int n_rxn = in_sizes[3];
    const int n_sub = in_sizes[16];

    float* agg = (float*)d_ws;        // N_RXN floats of scratch
    float* out = (float*)d_out;       // N_MET floats

    // 1) zero reaction accumulators
    zero_kernel<<<(n_rxn + 255) / 256, 256, 0, stream>>>(agg, n_rxn);

    // 2) edge-message MLP (WMMA) -> atomicAdd into agg
    {
        const int tiles  = (n_sub + 15) / 16;
        int blocks = (tiles + 63) / 64;            // ~8 tiles per wave
        if (blocks < 1) blocks = 1;
        mlp_kernel<0><<<blocks, 256, 0, stream>>>(x, nullptr, sto_all,
                                                  met_sub, sub_to_all, rxn_sub,
                                                  sW0, sb0, sW1, sb1, sW2, sb2,
                                                  agg, n_sub);
    }

    // 3) node homeostasis MLP (WMMA) -> seeds d_out
    {
        const int tiles  = (n_met + 15) / 16;
        int blocks = (tiles + 7) / 8;              // ~1 tile per wave
        if (blocks < 1) blocks = 1;
        mlp_kernel<1><<<blocks, 256, 0, stream>>>(x, a, nullptr,
                                                  nullptr, nullptr, nullptr,
                                                  nW0, nb0, nW1, nb1, nW2, nb2,
                                                  out, n_met);
    }

    // 4) v = 10^log_k * softplus(agg), in place
    rxn_kernel<<<(n_rxn + 255) / 256, 256, 0, stream>>>(log_k, agg, n_rxn);

    // 5) dxdt scatter-add on top of homeostasis
    contrib_kernel<<<(n_all + 255) / 256, 256, 0, stream>>>(sto_all, agg,
                                                            rxn_all, met_all,
                                                            out, n_all);
}
